// SVDQuantSegmentLinear_30150670418082
// MI455X (gfx1250) — 8-bit matrix operand order repaired and checked by reading, compiled, not run on hardware
//
#include <hip/hip_runtime.h>

typedef int v8i __attribute__((ext_vector_type(8)));
typedef int v4i __attribute__((ext_vector_type(4)));

typedef __attribute__((address_space(1))) v4i* gas_v4i_p;
typedef __attribute__((address_space(3))) v4i* las_v4i_p;

constexpr int Bb   = 4;
constexpr int Ss   = 1024;
constexpr int INK  = 4096;
constexpr int OUTN = 4096;
constexpr int CC   = 2;
constexpr int RR   = 32;
constexpr int Mtot = Bb * Ss;
constexpr int OCc  = OUTN / CC;

constexpr int BLK_M = 128, BLK_N = 128, BLK_K = 64;
constexpr int NKT   = INK / BLK_K;

#if defined(__has_builtin)
#  if __has_builtin(__builtin_amdgcn_global_load_async_to_lds_b128)
#    define HAVE_ASYNC_LDS 1
#  endif
#  if __has_builtin(__builtin_amdgcn_s_wait_asynccnt)
#    define HAVE_WAIT_ASYNC 1
#  endif
#endif

#if defined(HAVE_ASYNC_LDS)
template <int OFF>
__device__ __forceinline__ void async_copy16(const void* g, void* l) {
    __builtin_amdgcn_global_load_async_to_lds_b128(
        (gas_v4i_p)(g), (las_v4i_p)(l),  OFF,  0);
}
template <int N>
__device__ __forceinline__ void wait_asynccnt() {
#if defined(HAVE_WAIT_ASYNC)
    __builtin_amdgcn_s_wait_asynccnt(N);
#else
    asm volatile("s_wait_asynccnt %0" :: "i"(N) : "memory");
#endif
}
#endif

__device__ __forceinline__ int quant_i8(float v, float inv_scale) {
    int q = (int)rintf(v * inv_scale);
    q = q > 127 ? 127 : q;
    q = q < -127 ? -127 : q;
    return q;
}

__global__ __launch_bounds__(256)
void k_quant_x_low(const float* __restrict__ x, const float* __restrict__ smooth,
                   const float* __restrict__ l1, const float* __restrict__ in_scale,
                   signed char* __restrict__ qx, float* __restrict__ lowbuf)
{
    const int m = blockIdx.x;
    const int c = blockIdx.y;
    const int t = threadIdx.x;
    const float inv = 1.0f / in_scale[c];

    const float4* x4  = (const float4*)(x + (size_t)m * INK);
    const float4* sm4 = (const float4*)(smooth + (size_t)c * INK);
    int* qrow = (int*)(qx + ((size_t)c * Mtot + m) * INK);
    const float* l1c = l1 + (size_t)c * INK * RR;

    float acc[RR];
#pragma unroll
    for (int r = 0; r < RR; ++r) acc[r] = 0.f;

#pragma unroll
    for (int j = 0; j < 4; ++j) {
        const int idx = t + 256 * j;
        float4 xv = x4[idx];
        float4 sv = sm4[idx];
        float xs[4] = { xv.x / sv.x, xv.y / sv.y, xv.z / sv.z, xv.w / sv.w };

        int q0 = quant_i8(xs[0], inv), q1 = quant_i8(xs[1], inv);
        int q2 = quant_i8(xs[2], inv), q3 = quant_i8(xs[3], inv);
        qrow[idx] = (q0 & 0xff) | ((q1 & 0xff) << 8) | ((q2 & 0xff) << 16) | ((q3 & 0xff) << 24);

        const int kbase = idx * 4;
#pragma unroll
        for (int e = 0; e < 4; ++e) {
            const float4* lr = (const float4*)(l1c + (size_t)(kbase + e) * RR);
            const float xe = xs[e];
#pragma unroll
            for (int r4 = 0; r4 < 8; ++r4) {
                float4 lv = lr[r4];
                acc[r4 * 4 + 0] += xe * lv.x;
                acc[r4 * 4 + 1] += xe * lv.y;
                acc[r4 * 4 + 2] += xe * lv.z;
                acc[r4 * 4 + 3] += xe * lv.w;
            }
        }
    }

    __shared__ float sPart[8 * RR];
    const int lane = t & 31, wave = t >> 5;
#pragma unroll
    for (int r = 0; r < RR; ++r) {
        float v = acc[r];
        v += __shfl_down(v, 16, 32);
        v += __shfl_down(v, 8, 32);
        v += __shfl_down(v, 4, 32);
        v += __shfl_down(v, 2, 32);
        v += __shfl_down(v, 1, 32);
        if (lane == 0) sPart[wave * RR + r] = v;
    }
    __syncthreads();
    if (t < RR) {
        float s = 0.f;
#pragma unroll
        for (int w = 0; w < 8; ++w) s += sPart[w * RR + t];
        lowbuf[((size_t)c * Mtot + m) * RR + t] = s;
    }
}

__global__ __launch_bounds__(256)
void k_quant_w(const float* __restrict__ w, const float* __restrict__ w_scale,
               signed char* __restrict__ qw)
{
    const int idx = blockIdx.x * 256 + threadIdx.x;
    constexpr int perC = OCc * INK / 4;
    const int c = idx / perC;
    const float inv = 1.0f / w_scale[c];
    float4 wv = ((const float4*)w)[idx];
    int q0 = quant_i8(wv.x, inv), q1 = quant_i8(wv.y, inv);
    int q2 = quant_i8(wv.z, inv), q3 = quant_i8(wv.w, inv);
    ((int*)qw)[idx] = (q0 & 0xff) | ((q1 & 0xff) << 8) | ((q2 & 0xff) << 16) | ((q3 & 0xff) << 24);
}

__global__ __launch_bounds__(256)
void k_gemm_iu8(const signed char* __restrict__ qx, const signed char* __restrict__ qw,
                const float* __restrict__ lowbuf, const float* __restrict__ l2,
                const float* __restrict__ in_scale, const float* __restrict__ w_scale,
                const float* __restrict__ bias, float* __restrict__ out)
{
    const int c  = blockIdx.z;
    const int tm = blockIdx.y;
    const int tn = blockIdx.x;
    const int t  = threadIdx.x;
    const int lane = t & 31, wave = t >> 5;
    const int waveM = wave >> 2;
    const int waveN = wave & 3;

    __shared__ int4  sA4[2][BLK_M * BLK_K / 16];
    __shared__ int4  sB4[2][BLK_N * BLK_K / 16];
    __shared__ float sLow[BLK_M * RR];
    __shared__ float sL2[RR * BLK_N];

    {
        const float4* g = (const float4*)(lowbuf + ((size_t)c * Mtot + (size_t)tm * BLK_M) * RR);
        float4* d = (float4*)sLow;
#pragma unroll
        for (int p = 0; p < 4; ++p) d[t + 256 * p] = g[t + 256 * p];

        const float* l2c = l2 + (size_t)c * RR * OCc + (size_t)tn * BLK_N;
        float4* d2 = (float4*)sL2;
#pragma unroll
        for (int p = 0; p < 4; ++p) {
            const int i4 = t + 256 * p;
            const int r  = i4 >> 5;
            const int j4 = i4 & 31;
            d2[r * 32 + j4] = ((const float4*)(l2c + (size_t)r * OCc))[j4];
        }
    }

    const signed char* gAbase = qx + ((size_t)c * Mtot + (size_t)tm * BLK_M) * INK;
    const signed char* gBbase = qw + ((size_t)c * OCc + (size_t)tn * BLK_N) * INK;

    int laneOff[2];
#pragma unroll
    for (int p = 0; p < 2; ++p) {
        const int chunk = t + 256 * p;
        laneOff[p] = (chunk >> 2) * INK + (chunk & 3) * 16;
    }

    v8i acc[4][2];
#pragma unroll
    for (int mt = 0; mt < 4; ++mt)
#pragma unroll
        for (int nt = 0; nt < 2; ++nt)
            acc[mt][nt] = (v8i){0, 0, 0, 0, 0, 0, 0, 0};

    const int half = lane >> 4;
    const int l16  = lane & 15;
    const int aOff = (waveM * 64 + l16) * BLK_K + half * 16;
    const int bOff = (waveN * 32 + l16) * BLK_K + half * 16;

    auto compute_tile = [&](const int4* bufA, const int4* bufB) {
        const signed char* sA = (const signed char*)bufA + aOff;
        const signed char* sB = (const signed char*)bufB + bOff;
        v8i bfrag[2];
#pragma unroll
        for (int nt = 0; nt < 2; ++nt) {
            const signed char* bp = sB + nt * 1024;
            int4 r0 = *(const int4*)(bp);
            int4 r1 = *(const int4*)(bp + 32);
            bfrag[nt] = (v8i){r0.x, r0.y, r0.z, r0.w, r1.x, r1.y, r1.z, r1.w};
        }
#pragma unroll
        for (int mt = 0; mt < 4; ++mt) {
            const signed char* ap = sA + mt * 1024;
            int2 q0 = *(const int2*)(ap);
            int2 q1 = *(const int2*)(ap + 8);
            int2 q2 = *(const int2*)(ap + 32);
            int2 q3 = *(const int2*)(ap + 40);
            v8i afrag = (v8i){q0.x, q0.y, q1.x, q1.y, q2.x, q2.y, q3.x, q3.y};
#pragma unroll
            for (int nt = 0; nt < 2; ++nt) {
                acc[mt][nt] = __builtin_amdgcn_wmma_i32_16x16x64_iu8(
                    true, afrag, true, bfrag[nt], acc[mt][nt], false, false);
            }
        }
    };

#if defined(HAVE_ASYNC_LDS)
    const signed char* aP0 = gAbase + laneOff[0];
    const signed char* aP1 = gAbase + laneOff[1];
    const signed char* bP0 = gBbase + laneOff[0];
    const signed char* bP1 = gBbase + laneOff[1];

    async_copy16<0>(aP0, &sA4[0][t]);
    async_copy16<0>(aP1, &sA4[0][t + 256]);
    async_copy16<0>(bP0, &sB4[0][t]);
    async_copy16<0>(bP1, &sB4[0][t + 256]);
    async_copy16<BLK_K>(aP0, &sA4[1][t]);
    async_copy16<BLK_K>(aP1, &sA4[1][t + 256]);
    async_copy16<BLK_K>(bP0, &sB4[1][t]);
    async_copy16<BLK_K>(bP1, &sB4[1][t + 256]);

    for (int kt = 0; kt < NKT - 2; kt += 2) {
        aP0 += 2 * BLK_K; aP1 += 2 * BLK_K;
        bP0 += 2 * BLK_K; bP1 += 2 * BLK_K;

        wait_asynccnt<4>();
        __syncthreads();
        compute_tile(sA4[0], sB4[0]);
        __syncthreads();
        async_copy16<0>(aP0, &sA4[0][t]);
        async_copy16<0>(aP1, &sA4[0][t + 256]);
        async_copy16<0>(bP0, &sB4[0][t]);
        async_copy16<0>(bP1, &sB4[0][t + 256]);

        wait_asynccnt<4>();
        __syncthreads();
        compute_tile(sA4[1], sB4[1]);
        __syncthreads();
        async_copy16<BLK_K>(aP0, &sA4[1][t]);
        async_copy16<BLK_K>(aP1, &sA4[1][t + 256]);
        async_copy16<BLK_K>(bP0, &sB4[1][t]);
        async_copy16<BLK_K>(bP1, &sB4[1][t + 256]);
    }
    wait_asynccnt<4>();
    __syncthreads();
    compute_tile(sA4[0], sB4[0]);
    wait_asynccnt<0>();
    __syncthreads();
    compute_tile(sA4[1], sB4[1]);
#else
    for (int kt = 0; kt < NKT; ++kt) {
        __syncthreads();
#pragma unroll
        for (int p = 0; p < 2; ++p) {
            sA4[0][t + 256 * p] = *(const int4*)(gAbase + laneOff[p] + kt * BLK_K);
            sB4[0][t + 256 * p] = *(const int4*)(gBbase + laneOff[p] + kt * BLK_K);
        }
        __syncthreads();
        compute_tile(sA4[0], sB4[0]);
    }
#endif

    const float scale = in_scale[c] * w_scale[c];
    const float* biasc = bias + c * OCc + tn * BLK_N;
    float* outBase = out + (size_t)(tm * BLK_M) * OUTN + c * OCc + tn * BLK_N;

#pragma unroll
    for (int mt = 0; mt < 4; ++mt) {
#pragma unroll
        for (int i = 0; i < 8; ++i) {
            const int row = waveM * 64 + mt * 16 + half * 8 + i;
            const float* lrow = sLow + row * RR;
#pragma unroll
            for (int nt = 0; nt < 2; ++nt) {
                const int col = waveN * 32 + nt * 16 + l16;
                float d = 0.f;
#pragma unroll
                for (int r = 0; r < RR; ++r) d += lrow[r] * sL2[r * BLK_N + col];
                outBase[(size_t)row * OUTN + col] =
                    scale * (float)acc[mt][nt][i] + d + biasc[col];
            }
        }
    }
}

extern "C" void kernel_launch(void* const* d_in, const int* in_sizes, int n_in,
                              void* d_out, int out_size, void* d_ws, size_t ws_size,
                              hipStream_t stream)
{
    const float* x        = (const float*)d_in[0];
    const float* weight   = (const float*)d_in[1];
    const float* l1       = (const float*)d_in[2];
    const float* l2       = (const float*)d_in[3];
    const float* smooth   = (const float*)d_in[4];
    const float* in_scale = (const float*)d_in[5];
    const float* w_scale  = (const float*)d_in[6];
    const float* bias     = (const float*)d_in[7];
    float* out = (float*)d_out;

    signed char* qx = (signed char*)d_ws;
    signed char* qw = qx + (size_t)CC * Mtot * INK;
    float* lowbuf   = (float*)(qw + (size_t)CC * OCc * INK);

    k_quant_x_low<<<dim3(Mtot, CC), 256, 0, stream>>>(x, smooth, l1, in_scale, qx, lowbuf);
    k_quant_w<<<dim3(CC * OCc * INK / 4 / 256), 256, 0, stream>>>(weight, w_scale, qw);
    k_gemm_iu8<<<dim3(OCc / BLK_N, Mtot / BLK_M, CC), 256, 0, stream>>>(
        qx, qw, lowbuf, l2, in_scale, w_scale, bias, out);
}
